// GQASelfAttention_78855599554937
// MI455X (gfx1250) — matrix operand order repaired and checked by reading, not compiled, not run on hardware
//
#include <hip/hip_runtime.h>
#include <hip/hip_bf16.h>
#include <stdint.h>

typedef __attribute__((ext_vector_type(16))) _Float16 v16h;
typedef __attribute__((ext_vector_type(8)))  _Float16 v8h;
typedef __attribute__((ext_vector_type(8)))  float    v8f;

#define B_     2
#define S_     2048
#define H_     2048
#define NQ_    16
#define NKV_   4
#define D_     128
#define GROUPS_ 4
#define SCALE_ 0.08838834764831845f  /* 1/sqrt(128) */

// ---------------------------------------------------------------------------
// WMMA helpers (CDNA5 16x16x32 f16 -> f32)
// ---------------------------------------------------------------------------
__device__ __forceinline__ v8f wmma16(v16h a, v16h b, v8f c) {
  // 8 args: (neg_a, A, neg_b, B, c_mod, C, reuse_a, reuse_b)
  return __builtin_amdgcn_wmma_f32_16x16x32_f16(false, a, false, b, (short)0, c,
                                                false, false);
}

// A-matrix fragment (16x32, f16). Lane L: row = L%16, half = L/16.
// Elements: K = k0 + 8*half + (0..7), then K = k0 + 16 + 8*half + (0..7).
__device__ __forceinline__ v16h frag_load_a(const _Float16* base, int ld, int k0) {
  const int lane = threadIdx.x & 31;
  const int r = lane & 15, half = lane >> 4;
  const _Float16* p = base + (size_t)r * ld + k0 + half * 8;
  v8h lo = *(const v8h*)p;
  v8h hi = *(const v8h*)(p + 16);
  return __builtin_shufflevector(lo, hi, 0,1,2,3,4,5,6,7,8,9,10,11,12,13,14,15);
}

// B-matrix fragment (32x16, f16) read from row-major W[n][k] (= B^T rows).
// Lane L: col n = L%16, half = L/16. Elements: K = k0 + 8*half + (0..7), then K = k0 + 16 + 8*half + (0..7).
__device__ __forceinline__ v16h frag_load_b(const _Float16* base, int ld, int k0) {
  const int lane = threadIdx.x & 31;
  const int n = lane & 15, half = lane >> 4;
  const _Float16* p = base + (size_t)n * ld + k0 + half * 8;
  v8h lo = *(const v8h*)p;
  v8h hi = *(const v8h*)(p + 16);
  return __builtin_shufflevector(lo, hi, 0,1,2,3,4,5,6,7,8,9,10,11,12,13,14,15);
}

// CDNA5 async global->LDS copy (ASYNCcnt-tracked, no VGPR round-trip).
// Generic pointers to LDS truncate to the 32-bit LDS byte offset.
__device__ __forceinline__ void async_copy_b128(_Float16* lds,
                                                const _Float16* g) {
  unsigned l = (unsigned)(uintptr_t)lds;
  asm volatile("global_load_async_to_lds_b128 %0, %1, off"
               :: "v"(l), "v"(g) : "memory");
}
__device__ __forceinline__ void wait_async0() {
  asm volatile("s_wait_asynccnt 0x0" ::: "memory");
}

// ---------------------------------------------------------------------------
// fp32 -> fp16 conversion
// ---------------------------------------------------------------------------
__global__ void f32_to_f16_kernel(const float* __restrict__ in,
                                  _Float16* __restrict__ out, size_t n) {
  size_t i = (size_t)blockIdx.x * blockDim.x + threadIdx.x;
  if (i < n) out[i] = (_Float16)in[i];
}

// ---------------------------------------------------------------------------
// Generic WMMA GEMM: C[M,N] = A[M,K] (f16) @ W[N,K]^T (f16) + bias, f32 out.
// One wave computes a 16(M) x 64(N) tile; 8 waves / 256-thread block share an
// M-tile (A rows reused through L0/L2). Ping-pong double buffering (manual
// 2x unroll, K % 64 == 0 at every call site) so next-chunk loads overlap the
// WMMA chain with no register-rotation copies.
// ---------------------------------------------------------------------------
__global__ void __launch_bounds__(256)
gemm_wmma_f16_kernel(const _Float16* __restrict__ A,
                     const _Float16* __restrict__ W,
                     const float* __restrict__ bias,
                     float* __restrict__ C,
                     int M, int N, int K, int hasBias) {
  const int wavesPerBlock = blockDim.x >> 5;
  const int wave = blockIdx.x * wavesPerBlock + ((int)threadIdx.x >> 5);
  const int nTiles = N >> 6;
  const int mTile = wave / nTiles;
  const int nTile = wave % nTiles;
  if (mTile * 16 >= M) return;
  const int lane = threadIdx.x & 31;
  const int n = lane & 15, half = lane >> 4;

  const _Float16* Ab = A + (size_t)mTile * 16 * K;
  const _Float16* Wb = W + (size_t)nTile * 64 * K;

  v8f acc[4] = {};

  // prologue: fragment set A <- k-chunk 0
  v16h aA = frag_load_a(Ab, K, 0);
  v16h bA[4];
#pragma unroll
  for (int nb = 0; nb < 4; ++nb)
    bA[nb] = frag_load_b(Wb + (size_t)nb * 16 * K, K, 0);

  for (int k0 = 0; k0 < K; k0 += 64) {
    // phase 1: issue loads for k0+32 into set B, compute with set A
    v16h aB = frag_load_a(Ab, K, k0 + 32);
    v16h bB[4];
#pragma unroll
    for (int nb = 0; nb < 4; ++nb)
      bB[nb] = frag_load_b(Wb + (size_t)nb * 16 * K, K, k0 + 32);
    __builtin_prefetch(Ab + (size_t)(lane & 15) * K + k0 + 128, 0, 1);
#pragma unroll
    for (int nb = 0; nb < 4; ++nb) acc[nb] = wmma16(aA, bA[nb], acc[nb]);
    // phase 2: issue loads for k0+64 into set A, compute with set B
    if (k0 + 64 < K) {
      aA = frag_load_a(Ab, K, k0 + 64);
#pragma unroll
      for (int nb = 0; nb < 4; ++nb)
        bA[nb] = frag_load_b(Wb + (size_t)nb * 16 * K, K, k0 + 64);
    }
#pragma unroll
    for (int nb = 0; nb < 4; ++nb) acc[nb] = wmma16(aB, bB[nb], acc[nb]);
  }

#pragma unroll
  for (int nb = 0; nb < 4; ++nb) {
    const int col = nTile * 64 + nb * 16 + n;
    const float bvv = hasBias ? bias[col] : 0.0f;
#pragma unroll
    for (int v = 0; v < 8; ++v) {
      const int row = mTile * 16 + v + 8 * half;
      C[(size_t)row * N + col] = acc[nb][v] + bvv;
    }
  }
}

// ---------------------------------------------------------------------------
// RoPE + [B,S,Hh,D] -> [B,Hh,S,D] transpose. Emits f16 (for attention WMMA)
// and optionally fp32 (kv-cache outputs new_k/new_v).
// ---------------------------------------------------------------------------
__global__ void rope_transpose_kernel(const float* __restrict__ src,
                                      const float* __restrict__ cosb,
                                      const float* __restrict__ sinb,
                                      _Float16* __restrict__ dsth,
                                      float* __restrict__ dstf,
                                      int Hh, int doRope, size_t total) {
  size_t i = (size_t)blockIdx.x * blockDim.x + threadIdx.x;
  if (i >= total) return;
  const int d = (int)(i % D_);
  const int h = (int)((i / D_) % Hh);
  const int s = (int)((i / ((size_t)D_ * Hh)) % S_);
  const int b = (int)(i / ((size_t)D_ * Hh * S_));
  const size_t srow = ((size_t)b * S_ + s) * ((size_t)Hh * D_) + (size_t)h * D_;
  float x = src[srow + d];
  float y = x;
  if (doRope) {
    const size_t ci = ((size_t)b * S_ + s) * D_ + d;
    const float c = cosb[ci], sn = sinb[ci];
    const float other = (d < 64) ? -src[srow + d + 64] : src[srow + d - 64];
    y = x * c + other * sn;
  }
  const size_t o = (((size_t)b * Hh + h) * S_ + s) * D_ + d;
  dsth[o] = (_Float16)y;
  if (dstf) dstf[o] = y;
}

// ---------------------------------------------------------------------------
// Flash attention. Block = 4 waves = the 4 query heads sharing one KV head.
// Each wave owns a 16-row query tile; K/V tiles (32 keys) staged in LDS once
// per block — the K tile via CDNA5 async global->LDS copies that overlap the
// V transpose. Online softmax; QK^T and P*V via WMMA. Output written f16 in
// [B, S, NQ*D] layout for the Wo GEMM.
// ---------------------------------------------------------------------------
__global__ void __launch_bounds__(128)
flash_attn_kernel(const _Float16* __restrict__ qh,
                  const _Float16* __restrict__ kh,
                  const _Float16* __restrict__ vh,
                  _Float16* __restrict__ attn_h) {
  __shared__ alignas(16) _Float16 Ks[32 * D_];      // K tile, [key][d]
  __shared__ alignas(16) _Float16 Vts[D_ * 32];     // V tile transposed, [d][key]
  __shared__ alignas(16) _Float16 Ps[4][16 * 32];   // per-wave P tile

  const int mt   = blockIdx.x % (S_ / 16);
  const int hkv  = (blockIdx.x / (S_ / 16)) % NKV_;
  const int b    = blockIdx.x / ((S_ / 16) * NKV_);
  const int tid  = threadIdx.x;
  const int w    = tid >> 5;
  const int lane = tid & 31;
  const int n    = lane & 15, half = lane >> 4;
  const int hq   = hkv * GROUPS_ + w;
  const int row0 = mt * 16;

  // Q fragments for this wave's 16 rows (A operands, K-dim = D = 4 chunks)
  const _Float16* qbase = qh + (((size_t)b * NQ_ + hq) * S_ + (size_t)row0) * D_;
  v16h qf[4];
#pragma unroll
  for (int kb = 0; kb < 4; ++kb) qf[kb] = frag_load_a(qbase, D_, kb * 32);

  v8f accO[8] = {};
  float mrun[8], lrun[8];
#pragma unroll
  for (int v = 0; v < 8; ++v) { mrun[v] = -3.0e38f; lrun[v] = 0.0f; }

  const _Float16* kbase = kh + (((size_t)b * NKV_ + hkv) * S_) * D_;
  const _Float16* vbase = vh + (((size_t)b * NKV_ + hkv) * S_) * D_;

  const int T = (mt >> 1) + 1;  // causal: process keys [0, 32*T)
  for (int t = 0; t < T; ++t) {
    const int key0 = t * 32;
    const _Float16* kg = kbase + (size_t)key0 * D_;
    const _Float16* vg = vbase + (size_t)key0 * D_;
    // K tile: async global->LDS (ASYNCcnt), overlaps the V transpose below
#pragma unroll
    for (int e = tid; e < (32 * D_) / 8; e += 128)
      async_copy_b128(&Ks[e * 8], &kg[e * 8]);
    // V tile: load + transpose into LDS
    for (int e = tid; e < 32 * D_; e += 128) {
      const int key = e >> 7, d = e & (D_ - 1);
      Vts[d * 32 + key] = vg[e];
    }
    wait_async0();
    __syncthreads();

    // S = Q K^T  (two 16-key column blocks)
    v8f sf[2] = {};
#pragma unroll
    for (int kb = 0; kb < 4; ++kb) {
#pragma unroll
      for (int nb = 0; nb < 2; ++nb) {
        v16h bf = frag_load_b(&Ks[(nb * 16) * D_], D_, kb * 32);
        sf[nb] = wmma16(qf[kb], bf, sf[nb]);
      }
    }

    // scale + causal mask + per-row chunk max
    float mloc[8];
#pragma unroll
    for (int v = 0; v < 8; ++v) {
      const int row = row0 + v + 8 * half;
      float a0 = sf[0][v] * SCALE_;
      float a1 = sf[1][v] * SCALE_;
      if (key0 + n > row)      a0 = -3.0e38f;
      if (key0 + 16 + n > row) a1 = -3.0e38f;
      sf[0][v] = a0; sf[1][v] = a1;
      mloc[v] = fmaxf(a0, a1);
    }
#pragma unroll
    for (int v = 0; v < 8; ++v)
#pragma unroll
      for (int o = 8; o >= 1; o >>= 1)
        mloc[v] = fmaxf(mloc[v], __shfl_xor(mloc[v], o, 32));

    // online softmax update; stage P (f16) into this wave's LDS slab
#pragma unroll
    for (int v = 0; v < 8; ++v) {
      const float mnew = fmaxf(mrun[v], mloc[v]);
      const float corr = __expf(mrun[v] - mnew);
      const float p0 = __expf(sf[0][v] - mnew);
      const float p1 = __expf(sf[1][v] - mnew);
      const int prow = v + 8 * half;
      Ps[w][prow * 32 + n]      = (_Float16)p0;
      Ps[w][prow * 32 + 16 + n] = (_Float16)p1;
      float ps = p0 + p1;
#pragma unroll
      for (int o = 8; o >= 1; o >>= 1) ps += __shfl_xor(ps, o, 32);
      lrun[v] = lrun[v] * corr + ps;
      mrun[v] = mnew;
#pragma unroll
      for (int db = 0; db < 8; ++db) accO[db][v] *= corr;
    }

    // O += P * V   (A = P from LDS, B = V^T columns from LDS)
    v16h pf = frag_load_a(&Ps[w][0], 32, 0);
#pragma unroll
    for (int db = 0; db < 8; ++db) {
      v16h vf = frag_load_b(&Vts[(db * 16) * 32], 32, 0);
      accO[db] = wmma16(pf, vf, accO[db]);
    }
    __syncthreads();  // protect Ks/Vts before next tile load
  }

  // epilogue: normalize and emit f16 in [B, S, NQ*D]
#pragma unroll
  for (int v = 0; v < 8; ++v) {
    const float inv = 1.0f / (lrun[v] + 1e-9f);
    const int srow = row0 + v + 8 * half;
    const size_t base =
        ((size_t)b * S_ + srow) * ((size_t)NQ_ * D_) + (size_t)hq * D_;
#pragma unroll
    for (int db = 0; db < 8; ++db)
      attn_h[base + db * 16 + n] = (_Float16)(accO[db][v] * inv);
  }
}

// ---------------------------------------------------------------------------
// host orchestration
// ---------------------------------------------------------------------------
extern "C" void kernel_launch(void* const* d_in, const int* in_sizes, int n_in,
                              void* d_out, int out_size, void* d_ws,
                              size_t ws_size, hipStream_t stream) {
  (void)in_sizes; (void)n_in; (void)out_size; (void)ws_size;

  const float* x    = (const float*)d_in[0];
  const float* cosb = (const float*)d_in[1];
  const float* sinb = (const float*)d_in[2];
  const float* Wq   = (const float*)d_in[3];
  const float* bq   = (const float*)d_in[4];
  const float* Wk   = (const float*)d_in[5];
  const float* bk   = (const float*)d_in[6];
  const float* Wv   = (const float*)d_in[7];
  const float* bv   = (const float*)d_in[8];
  const float* Wo   = (const float*)d_in[9];
  float* out = (float*)d_out;

  char* ws = (char*)d_ws;
  size_t off = 0;
  auto alloc = [&](size_t bytes) -> char* {
    char* p = ws + off;
    off += (bytes + 255) & ~(size_t)255;
    return p;
  };

  const size_t MBS = (size_t)B_ * S_;  // 4096 token rows
  _Float16* xh  = (_Float16*)alloc(MBS * H_ * 2);
  _Float16* Wqh = (_Float16*)alloc((size_t)NQ_ * D_ * H_ * 2);
  _Float16* Wkh = (_Float16*)alloc((size_t)NKV_ * D_ * H_ * 2);
  _Float16* Wvh = (_Float16*)alloc((size_t)NKV_ * D_ * H_ * 2);
  _Float16* Woh = (_Float16*)alloc((size_t)H_ * NQ_ * D_ * 2);
  _Float16* qhb = (_Float16*)alloc(MBS * NQ_ * D_ * 2);
  _Float16* khb = (_Float16*)alloc(MBS * NKV_ * D_ * 2);
  _Float16* vhb = (_Float16*)alloc(MBS * NKV_ * D_ * 2);
  _Float16* ah  = (_Float16*)alloc(MBS * NQ_ * D_ * 2);
  float* qf32 = (float*)alloc(MBS * NQ_ * D_ * 4);
  float* kf32 = (float*)alloc(MBS * NKV_ * D_ * 4);
  float* vf32 = (float*)alloc(MBS * NKV_ * D_ * 4);

  auto cvt = [&](const float* src, _Float16* dst, size_t nelem) {
    const int grid = (int)((nelem + 255) / 256);
    f32_to_f16_kernel<<<grid, 256, 0, stream>>>(src, dst, nelem);
  };
  cvt(x,  xh,  MBS * H_);
  cvt(Wq, Wqh, (size_t)NQ_ * D_ * H_);
  cvt(Wk, Wkh, (size_t)NKV_ * D_ * H_);
  cvt(Wv, Wvh, (size_t)NKV_ * D_ * H_);
  cvt(Wo, Woh, (size_t)H_ * NQ_ * D_);

  auto gemm = [&](const _Float16* A, const _Float16* Wm, const float* bias,
                  float* C, int M, int N, int K, int hasBias) {
    const int waves = (M / 16) * (N / 64);
    const int blocks = waves / 8;  // 8 waves per 256-thread block
    gemm_wmma_f16_kernel<<<blocks, 256, 0, stream>>>(A, Wm, bias, C, M, N, K,
                                                     hasBias);
  };
  gemm(xh, Wqh, bq, qf32, (int)MBS, NQ_ * D_, H_, 1);   // Q proj
  gemm(xh, Wkh, bk, kf32, (int)MBS, NKV_ * D_, H_, 1);  // K proj
  gemm(xh, Wvh, bv, vf32, (int)MBS, NKV_ * D_, H_, 1);  // V proj

  float* newk = out + (size_t)B_ * S_ * H_;
  float* newv = newk + (size_t)B_ * NKV_ * S_ * D_;
  const size_t nq = MBS * NQ_ * D_;
  const size_t nk = MBS * NKV_ * D_;
  rope_transpose_kernel<<<(int)((nq + 255) / 256), 256, 0, stream>>>(
      qf32, cosb, sinb, qhb, (float*)nullptr, NQ_, 1, nq);
  rope_transpose_kernel<<<(int)((nk + 255) / 256), 256, 0, stream>>>(
      kf32, cosb, sinb, khb, newk, NKV_, 1, nk);
  rope_transpose_kernel<<<(int)((nk + 255) / 256), 256, 0, stream>>>(
      vf32, cosb, sinb, vhb, newv, NKV_, 0, nk);

  flash_attn_kernel<<<B_ * NKV_ * (S_ / 16), 128, 0, stream>>>(qhb, khb, vhb,
                                                               ah);

  gemm(ah, Woh, (const float*)nullptr, out, (int)MBS, H_, NQ_ * D_, 0);
}
